// Model_43422119362736
// MI455X (gfx1250) — matrix operand order repaired and checked by reading, not compiled, not run on hardware
//
#include <hip/hip_runtime.h>
#include <stdint.h>

// LSTM: B=8192, T=512, IN=8, HID=64, OUT=2 (fp32 in/out, bf16 WMMA matmul)
// Per block: 16 batch rows, 4 waves. Wave w owns hid columns [16w,16w+16) of
// ALL FOUR gates -> pointwise LSTM update is fully in-register.
// A = [h(64) | x(8) | 1.0 | 0(23)]  (K=96 = 3x32)
// B = [W_hh ; W_ih ; bias ; 0]      -> bias folded into the WMMA, C starts at 0.

#define B_TOT    8192
#define T_LEN    512
#define IN_DIM   8
#define HID      64
#define TILE     16           // batch rows per block
#define NTHREADS 128          // 4 waves of 32
#define ASTRIDE  100          // bf16 elems per A row (even, padded vs bank conflicts)
#define HSTRIDE  68           // f32 elems per row of final-h buffer
#define KONE     72           // K index of the constant-1 column (bias row in B)

typedef __attribute__((ext_vector_type(16))) __bf16    v16bf;
typedef __attribute__((ext_vector_type(8)))  float     v8f;
typedef __attribute__((ext_vector_type(8)))  unsigned  v8u;

union FragAB { v8u u; v16bf bf; };

__device__ __forceinline__ unsigned short f2bf(float f) {
    union { float f; unsigned u; } x; x.f = f;
    unsigned r = x.u + 0x7FFFu + ((x.u >> 16) & 1u);   // RNE
    return (unsigned short)(r >> 16);
}
__device__ __forceinline__ unsigned packbf(float a, float b) {
    return (unsigned)f2bf(a) | ((unsigned)f2bf(b) << 16);
}

__device__ __forceinline__ float tanh_fast(float x) {
#if __has_builtin(__builtin_amdgcn_tanhf)
    return __builtin_amdgcn_tanhf(x);        // v_tanh_f32 (single TRANS op)
#else
    float e = __expf(2.0f * x);
    return 1.0f - 2.0f / (e + 1.0f);
#endif
}
__device__ __forceinline__ float sigmoid_fast(float x) {
    return fmaf(0.5f, tanh_fast(0.5f * x), 0.5f);
}

__global__ __launch_bounds__(NTHREADS)
void lstm_fused_kernel(const float* __restrict__ x,
                       const float* __restrict__ W_ih,   // [256, 8]
                       const float* __restrict__ W_hh,   // [256, 64]
                       const float* __restrict__ b_ih,   // [256]
                       const float* __restrict__ b_hh,   // [256]
                       const float* __restrict__ W_fc,   // [2, 64]
                       const float* __restrict__ b_fc,   // [2]
                       float* __restrict__ out)          // [8192, 2]
{
    // double-buffered A tile: [h(0..63) | x(64..71) | 1.0(72) | 0(73..95)] bf16
    __shared__ unsigned short aLds[2][TILE * ASTRIDE];
    __shared__ float          hLds[TILE * HSTRIDE];      // f32 h at t = T-1

    const int tid   = threadIdx.x;
    const int lane  = tid & 31;
    const int wave  = tid >> 5;
    const int n     = lane & 15;
    const int jcol  = wave * 16 + n;         // this lane's hidden-unit index
    const int rbase = (lane >> 4) << 3;      // C/D rows: rbase .. rbase+7
    const int b0    = blockIdx.x * TILE;

    // ---- one-time init ------------------------------------------------------
    {
        unsigned short* az = &aLds[0][0];
        for (int i = tid; i < 2 * TILE * ASTRIDE; i += NTHREADS) az[i] = 0;
    }
    __syncthreads();
    // constant-1 column at K=72 in both buffers (never overwritten later)
    if (tid < TILE * 2) {
        int buf = tid & 1, row = tid >> 1;
        aLds[buf][row * ASTRIDE + KONE] = f2bf(1.0f);
    }

    float creg[8];                            // c-state: rows rbase..rbase+7, col jcol
    #pragma unroll
    for (int r = 0; r < 8; ++r) creg[r] = 0.0f;

    // B fragments: Wcat[k][col] = k<64 ? W_hh[col][k] : k<72 ? W_ih[col][k-64]
    //                           : k==72 ? (b_ih[col]+b_hh[col]) : 0
    // B k order used here (contiguous, same as A): lane col = lane&15; lanes>=16 use K+16; VGPR v packs K=2v,2v+1.
    FragAB bfrag[4][3];
    #pragma unroll
    for (int g = 0; g < 4; ++g) {
        int col = g * HID + jcol;
        float bsum = b_ih[col] + b_hh[col];
        #pragma unroll
        for (int c = 0; c < 3; ++c) {
            #pragma unroll
            for (int v = 0; v < 8; ++v) {
                int k  = c * 32 + ((lane >> 4) << 4) + 2 * v;
                int k1 = k + 1;
                float e0 = (k  < 64) ? W_hh[col * 64 + k]
                         : (k  < 72) ? W_ih[col * 8 + (k - 64)]
                         : (k  == KONE) ? bsum : 0.0f;
                float e1 = (k1 < 64) ? W_hh[col * 64 + k1]
                         : (k1 < 72) ? W_ih[col * 8 + (k1 - 64)]
                         : (k1 == KONE) ? bsum : 0.0f;
                bfrag[g][c].u[v] = packbf(e0, e1);
            }
        }
    }
    __syncthreads();

    const v8f zeroC = (v8f)(0.0f);            // loop-invariant zero C matrix

    // ---- timestep loop (one barrier per step) -------------------------------
    for (int t = 0; t < T_LEN; ++t) {
        // x_t -> current buffer, cols 64..71 (one float per thread)
        {
            int row = tid >> 3, i = tid & 7;
            float xv = x[((size_t)(b0 + row) * T_LEN + t) * IN_DIM + i];
            aLds[t & 1][row * ASTRIDE + HID + i] = f2bf(xv);
        }
        __syncthreads();   // orders: prev-step h stores + x_t stores vs A reads below

        const unsigned short* abuf = aLds[t & 1];
        unsigned short*       hbuf = aLds[(t + 1) & 1];   // h_{t+1} -> other buffer

        // A fragments, contiguous k order matching B (k = 16*(lane>>4) + 2v, 2v+1)
        FragAB afrag[3];
        {
            const int rowA = lane & 15;
            const int khi  = (lane >> 4) ? 16 : 0;
            #pragma unroll
            for (int c = 0; c < 3; ++c) {
                #pragma unroll
                for (int v = 0; v < 8; ++v) {
                    int k = c * 32 + khi + 2 * v;
                    afrag[c].u[v] = *(const unsigned*)(&abuf[rowA * ASTRIDE + k]);
                }
            }
        }

        // 12 WMMAs: acc[g] = gate-g pre-activation (bias enters via K=72 chunk)
        v8f acc[4];
        #pragma unroll
        for (int g = 0; g < 4; ++g) {
            acc[g] = __builtin_amdgcn_wmma_f32_16x16x32_bf16(
                        false, afrag[2].bf, false, bfrag[g][2].bf,
                        (short)0, zeroC, false, false);
            acc[g] = __builtin_amdgcn_wmma_f32_16x16x32_bf16(
                        false, afrag[0].bf, false, bfrag[g][0].bf,
                        (short)0, acc[g], false, false);
            acc[g] = __builtin_amdgcn_wmma_f32_16x16x32_bf16(
                        false, afrag[1].bf, false, bfrag[g][1].bf,
                        (short)0, acc[g], false, false);
        }

        // pointwise LSTM update, fully in registers (torch gate order i,f,g,o)
        #pragma unroll
        for (int r = 0; r < 8; ++r) {
            float iv = sigmoid_fast(acc[0][r]);
            float fv = sigmoid_fast(acc[1][r]);
            float gv = tanh_fast   (acc[2][r]);
            float ov = sigmoid_fast(acc[3][r]);
            float cN = fmaf(fv, creg[r], iv * gv);
            creg[r]  = cN;
            float hN = ov * tanh_fast(cN);
            hbuf[(rbase + r) * ASTRIDE + jcol] = f2bf(hN);
            if (t == T_LEN - 1)
                hLds[(rbase + r) * HSTRIDE + jcol] = hN;   // f32 h for final FC
        }
    }
    __syncthreads();

    // ---- final FC: out[b, o] = h . W_fc[o] + b_fc[o] ------------------------
    if (tid < TILE * 2) {
        int row = tid >> 1, o = tid & 1;
        float sum = b_fc[o];
        #pragma unroll 8
        for (int j = 0; j < HID; ++j)
            sum += hLds[row * HSTRIDE + j] * W_fc[o * HID + j];
        out[(size_t)(b0 + row) * 2 + o] = sum;
    }
}

extern "C" void kernel_launch(void* const* d_in, const int* in_sizes, int n_in,
                              void* d_out, int out_size, void* d_ws, size_t ws_size,
                              hipStream_t stream) {
    const float* x    = (const float*)d_in[0];
    const float* W_ih = (const float*)d_in[1];
    const float* W_hh = (const float*)d_in[2];
    const float* b_ih = (const float*)d_in[3];
    const float* b_hh = (const float*)d_in[4];
    const float* W_fc = (const float*)d_in[5];
    const float* b_fc = (const float*)d_in[6];
    float* out = (float*)d_out;

    dim3 grid(B_TOT / TILE);   // 512 blocks
    dim3 block(NTHREADS);      // 4 waves
    lstm_fused_kernel<<<grid, block, 0, stream>>>(x, W_ih, W_hh, b_ih, b_hh,
                                                  W_fc, b_fc, out);
}
